// BinarizedLeNet5_BN_CIM_80917183857094
// MI455X (gfx1250) — 8-bit matrix operand order repaired and checked by reading, compiled, not run on hardware
//
#include <hip/hip_runtime.h>

typedef int   v16i __attribute__((ext_vector_type(16)));
typedef float v8f  __attribute__((ext_vector_type(8)));

#define ACT1_OFF    0u
#define ACT1_BYTES  (8192u*864u)
#define FC1IN_OFF   (ACT1_OFF + ACT1_BYTES)
#define FC1IN_BYTES (8192u*256u)
#define CONST_OFF   (FC1IN_OFF + FC1IN_BYTES)
#define CONST_BYTES 4096u
#define BPACK_OFF   (CONST_OFF + CONST_BYTES)

#define C_SC1 150
#define C_SH1 156
#define C_SC2 162
#define C_SH2 178
#define C_SCF1 194
#define C_SHF1 314
#define C_SCF2 434
#define C_SHF2 518
#define C_SCF3 602
#define C_SHF3 612

#define BW_CONV2 0
#define BW_FC1   1536
#define BW_FC2   9728
#define BW_FC3   12800
#define BW_C1BITS 13312

__device__ __forceinline__ unsigned char sgn8(float x) {
  return x >= 0.0f ? (unsigned char)0x38 : (unsigned char)0xB8;
}
__device__ __forceinline__ float clamp1(float v) { return fminf(fmaxf(v, -1.0f), 1.0f); }

__device__ __forceinline__ int kkA0(int v) {
  return ((v & 7) >> 1) * 16 + (v & 1) * 4 + (v >> 3) * 64;
}

__device__ __forceinline__ v16i load_btile(const unsigned int* __restrict__ tw, int lane) {
  const uint4* p = (const uint4*)tw + lane * 4;
  uint4 q0 = p[0], q1 = p[1], q2 = p[2], q3 = p[3];
  v16i b;
  b[0]=(int)q0.x; b[1]=(int)q0.y; b[2]=(int)q0.z; b[3]=(int)q0.w;
  b[4]=(int)q1.x; b[5]=(int)q1.y; b[6]=(int)q1.z; b[7]=(int)q1.w;
  b[8]=(int)q2.x; b[9]=(int)q2.y; b[10]=(int)q2.z; b[11]=(int)q2.w;
  b[12]=(int)q3.x; b[13]=(int)q3.y; b[14]=(int)q3.z; b[15]=(int)q3.w;
  return b;
}

__device__ __forceinline__ unsigned int packB(const float* __restrict__ W, int K, int N,
                                              int tileK, int tileN, int widx) {
  int lane = widx >> 4, v = widx & 15;
  int half = lane >> 4, nl = lane & 15;
  int n = tileN * 16 + nl;
  unsigned int wv = 0;
  for (int b = 0; b < 4; b++) {
    int kk = (v >> 3) * 64 + ((v >> 1) & 3) * 16 + (v & 1) * 4 + half * 8 + b;
    int k = tileK * 128 + kk;
    unsigned int by = 0;
    if (k < K && n < N) by = (W[n * K + k] >= 0.0f) ? 0x38u : 0xB8u;
    wv |= by << (8 * b);
  }
  return wv;
}

__device__ __forceinline__ unsigned int packB_conv2(const float* __restrict__ W,
                                                    int tileK, int widx) {
  int lane = widx >> 4, v = widx & 15;
  int half = lane >> 4, n = lane & 15;
  unsigned int wv = 0;
  for (int b = 0; b < 4; b++) {
    int kk = (v >> 3) * 64 + ((v >> 1) & 3) * 16 + (v & 1) * 4 + half * 8 + b;
    int k = tileK * 128 + kk;
    int dy = k / 72, rem = k - dy * 72;
    int dx = rem / 6, c = rem - dx * 6;
    unsigned int by = 0;
    if (dy < 5 && dx < 5)
      by = (W[n * 150 + c * 25 + dy * 5 + dx] >= 0.0f) ? 0x38u : 0xB8u;
    wv |= by << (8 * b);
  }
  return wv;
}

__global__ void prep_kernel(
    const float* c1w, const float* c1b, const float* c2w, const float* c2b,
    const float* f1w, const float* f1b, const float* f2w, const float* f2b,
    const float* f3w, const float* f3b,
    const float* b1g, const float* b1b, const float* b1m, const float* b1v,
    const float* b2g, const float* b2b, const float* b2m, const float* b2v,
    const float* g1g, const float* g1b, const float* g1m, const float* g1v,
    const float* g2g, const float* g2b, const float* g2m, const float* g2v,
    const float* g3g, const float* g3b, const float* g3m, const float* g3v,
    float* cst, unsigned int* bw)
{
  int g  = blockIdx.x * blockDim.x + threadIdx.x;
  int gs = gridDim.x * blockDim.x;
  for (int i = g; i < 6; i += gs) {
    float s = b1g[i] * rsqrtf(b1v[i] + 1e-5f);
    cst[C_SC1 + i] = s; cst[C_SH1 + i] = c1b[i] * s + b1b[i] - b1m[i] * s;
    unsigned int m = 0;
    for (int k = 0; k < 25; k++) m |= (c1w[i * 25 + k] >= 0.0f ? 1u : 0u) << k;
    bw[BW_C1BITS + i] = m;
  }
  for (int i = g; i < 16; i += gs) {
    float s = b2g[i] * rsqrtf(b2v[i] + 1e-5f);
    cst[C_SC2 + i] = s; cst[C_SH2 + i] = c2b[i] * s + b2b[i] - b2m[i] * s;
  }
  for (int i = g; i < 120; i += gs) {
    float s = g1g[i] * rsqrtf(g1v[i] + 1e-5f);
    cst[C_SCF1 + i] = s; cst[C_SHF1 + i] = f1b[i] * s + g1b[i] - g1m[i] * s;
  }
  for (int i = g; i < 84; i += gs) {
    float s = g2g[i] * rsqrtf(g2v[i] + 1e-5f);
    cst[C_SCF2 + i] = s; cst[C_SHF2 + i] = f2b[i] * s + g2b[i] - g2m[i] * s;
  }
  for (int i = g; i < 10; i += gs) {
    float s = g3g[i] * rsqrtf(g3v[i] + 1e-5f);
    cst[C_SCF3 + i] = s; cst[C_SHF3 + i] = f3b[i] * s + g3b[i] - g3m[i] * s;
  }
  for (int i = g; i < 1536; i += gs)
    bw[BW_CONV2 + i] = packB_conv2(c2w, i >> 9, i & 511);
  for (int i = g; i < 8192; i += gs) {
    int t = i >> 9;
    bw[BW_FC1 + i] = packB(f1w, 256, 120, t & 1, t >> 1, i & 511);
  }
  for (int i = g; i < 3072; i += gs)
    bw[BW_FC2 + i] = packB(f2w, 120, 84, 0, i >> 9, i & 511);
  for (int i = g; i < 512; i += gs)
    bw[BW_FC3 + i] = packB(f3w, 84, 10, 0, 0, i);
}

__global__ __launch_bounds__(128) void conv1_kernel(
    const float* __restrict__ x, const float* __restrict__ cst,
    const unsigned int* __restrict__ bwords, unsigned char* __restrict__ act1)
{
  __shared__ unsigned int swb[6];
  __shared__ float sA[6], sB[6];
  int tid = threadIdx.x;
  if (tid < 6) {
    swb[tid] = bwords[BW_C1BITS + tid];
    float s = cst[C_SC1 + tid];
    sA[tid] = -2.0f * s;
    sB[tid] = 25.0f * s + cst[C_SH1 + tid];
  }
  __syncthreads();

  int idx = blockIdx.x * 128 + tid;
  int b = idx / 144, rem = idx - b * 144;
  int y2 = rem / 12, x2 = rem - y2 * 12;

  const float* xim = x + (size_t)b * 784 + (2 * y2) * 28 + 2 * x2;
  unsigned int rowb[6];
  #pragma unroll
  for (int r = 0; r < 6; r++) {
    unsigned int m = 0;
    #pragma unroll
    for (int c = 0; c < 6; c++)
      m |= (xim[r * 28 + c] >= 0.0f ? 1u : 0u) << c;
    rowb[r] = m;
  }
  unsigned int wm[4];
  #pragma unroll
  for (int dy = 0; dy < 2; dy++)
    #pragma unroll
    for (int dx = 0; dx < 2; dx++) {
      unsigned int m = 0;
      #pragma unroll
      for (int i = 0; i < 5; i++)
        m |= ((rowb[i + dy] >> dx) & 31u) << (5 * i);
      wm[dy * 2 + dx] = m;
    }

  unsigned char* dst = act1 + (size_t)b * 864 + (y2 * 12 + x2) * 6;
  #pragma unroll
  for (int ch = 0; ch < 6; ch++) {
    unsigned int wb = swb[ch];
    float a = sA[ch], bb = sB[ch];
    float p = 0.0f;
    #pragma unroll
    for (int q = 0; q < 4; q++) {
      float t = (float)__popc(wm[q] ^ wb);
      p += clamp1(fmaf(t, a, bb));
    }
    dst[ch] = sgn8(p);
  }
}

__global__ __launch_bounds__(128) void conv2_kernel(
    const unsigned char* __restrict__ act1, const unsigned int* __restrict__ bw,
    const float* __restrict__ cst, unsigned char* __restrict__ fc1_in)
{
  __shared__ uint4 simg4[64];
  __shared__ float sc[16], sh[16];
  unsigned char* simg = (unsigned char*)simg4;
  int tid = threadIdx.x, b = blockIdx.x;
  if (tid < 64) {
    uint4 z; z.x = 0u; z.y = 0u; z.z = 0u; z.w = 0u;
    simg4[tid] = (tid < 54) ? ((const uint4*)(act1 + (size_t)b * 864))[tid] : z;
  }
  if (tid < 16) { sc[tid] = cst[C_SC2 + tid]; sh[tid] = cst[C_SH2 + tid]; }
  __syncthreads();

  int w = tid >> 5, lane = tid & 31, half = lane >> 4, mr = lane & 15;
  int q  = w * 4 + (mr >> 2);
  int ho = 2 * (q >> 2) + ((mr >> 1) & 1);
  int wo = 2 * (q & 3) + (mr & 1);

  const unsigned char* sp = simg + (ho * 72 + wo * 6 + half * 8);

  v8f acc = {0, 0, 0, 0, 0, 0, 0, 0};
  #pragma unroll
  for (int ks = 0; ks < 3; ks++) {
    v16i A;
    #pragma unroll
    for (int v = 0; v < 16; v++) {
      int o = ks * 128 + kkA0(v);
      unsigned int b0 = sp[o + 0];
      unsigned int b1 = sp[o + 1];
      unsigned int b2 = sp[o + 2];
      unsigned int b3 = sp[o + 3];
      A[v] = (int)(b0 | (b1 << 8) | (b2 << 16) | (b3 << 24));
    }
    v16i B = load_btile(bw + BW_CONV2 + ks * 512, lane);
    acc = __builtin_amdgcn_wmma_f32_16x16x128_fp8_fp8(A, B, (short)0, acc, false, false);
  }

  int n = mr;
  float p0 = 0.f, p1 = 0.f;
  #pragma unroll
  for (int j = 0; j < 4; j++) p0 += clamp1(acc[j] * sc[n] + sh[n]);
  #pragma unroll
  for (int j = 4; j < 8; j++) p1 += clamp1(acc[j] * sc[n] + sh[n]);
  int q0 = w * 4 + 2 * half;
  unsigned char* dst = fc1_in + (size_t)b * 256 + n * 16;
  dst[q0]     = sgn8(p0);
  dst[q0 + 1] = sgn8(p1);
}

__global__ __launch_bounds__(128) void fc_kernel(
    const unsigned char* __restrict__ fc1_in, const unsigned int* __restrict__ bw,
    const float* __restrict__ cst, float* __restrict__ out)
{
  __shared__ unsigned char a2[4][2048];
  __shared__ unsigned char a3[4][2048];
  int tid = threadIdx.x;
  {
    unsigned int* z2 = (unsigned int*)a2;
    unsigned int* z3 = (unsigned int*)a3;
    for (int i = tid; i < 2048; i += 128) { z2[i] = 0u; z3[i] = 0u; }
  }
  __syncthreads();

  int w = tid >> 5, lane = tid & 31, half = lane >> 4, mr = lane & 15;
  int rw = blockIdx.x * 64 + w * 16;

  const unsigned char* arow = fc1_in + (size_t)(rw + mr) * 256;
  v16i A0, A1;
  #pragma unroll
  for (int v = 0; v < 16; v++) {
    A0[v] = *(const int*)(arow + kkA0(v) + half * 8);
    A1[v] = *(const int*)(arow + 128 + kkA0(v) + half * 8);
  }
  unsigned char* w2 = &a2[w][0];
  #pragma unroll
  for (int nt = 0; nt < 8; nt++) {
    v8f c = {0, 0, 0, 0, 0, 0, 0, 0};
    v16i B0 = load_btile(bw + BW_FC1 + (nt * 2 + 0) * 512, lane);
    c = __builtin_amdgcn_wmma_f32_16x16x128_fp8_fp8(A0, B0, (short)0, c, false, false);
    v16i B1 = load_btile(bw + BW_FC1 + (nt * 2 + 1) * 512, lane);
    c = __builtin_amdgcn_wmma_f32_16x16x128_fp8_fp8(A1, B1, (short)0, c, false, false);
    int n = nt * 16 + mr;
    float scl = (n < 120) ? cst[C_SCF1 + n] : 0.f;
    float sft = (n < 120) ? cst[C_SHF1 + n] : 0.f;
    #pragma unroll
    for (int j = 0; j < 8; j++) {
      int row = half * 8 + j;
      w2[row * 128 + n] = (n < 120) ? sgn8(clamp1(c[j] * scl + sft)) : (unsigned char)0;
    }
  }
  __syncthreads();

  v16i A2;
  #pragma unroll
  for (int v = 0; v < 16; v++)
    A2[v] = *(const int*)(&a2[w][mr * 128 + kkA0(v) + half * 8]);
  unsigned char* w3 = &a3[w][0];
  #pragma unroll
  for (int nt = 0; nt < 6; nt++) {
    v8f c = {0, 0, 0, 0, 0, 0, 0, 0};
    v16i B = load_btile(bw + BW_FC2 + nt * 512, lane);
    c = __builtin_amdgcn_wmma_f32_16x16x128_fp8_fp8(A2, B, (short)0, c, false, false);
    int n = nt * 16 + mr;
    float scl = (n < 84) ? cst[C_SCF2 + n] : 0.f;
    float sft = (n < 84) ? cst[C_SHF2 + n] : 0.f;
    #pragma unroll
    for (int j = 0; j < 8; j++) {
      int row = half * 8 + j;
      w3[row * 128 + n] = (n < 84) ? sgn8(clamp1(c[j] * scl + sft)) : (unsigned char)0;
    }
  }
  __syncthreads();

  v16i A3;
  #pragma unroll
  for (int v = 0; v < 16; v++)
    A3[v] = *(const int*)(&a3[w][mr * 128 + kkA0(v) + half * 8]);
  v8f c = {0, 0, 0, 0, 0, 0, 0, 0};
  v16i B = load_btile(bw + BW_FC3, lane);
  c = __builtin_amdgcn_wmma_f32_16x16x128_fp8_fp8(A3, B, (short)0, c, false, false);
  int n = mr;
  if (n < 10) {
    float scl = cst[C_SCF3 + n], sft = cst[C_SHF3 + n];
    #pragma unroll
    for (int j = 0; j < 8; j++) {
      int row = rw + half * 8 + j;
      out[(size_t)row * 10 + n] = c[j] * scl + sft;
    }
  }
}

extern "C" void kernel_launch(void* const* d_in, const int* in_sizes, int n_in,
                              void* d_out, int out_size, void* d_ws, size_t ws_size,
                              hipStream_t stream) {
  (void)in_sizes; (void)n_in; (void)out_size; (void)ws_size;
  const float* x    = (const float*)d_in[0];
  const float* c1w  = (const float*)d_in[1];
  const float* c1b  = (const float*)d_in[2];
  const float* c2w  = (const float*)d_in[3];
  const float* c2b  = (const float*)d_in[4];
  const float* f1w  = (const float*)d_in[5];
  const float* f1b  = (const float*)d_in[6];
  const float* f2w  = (const float*)d_in[7];
  const float* f2b  = (const float*)d_in[8];
  const float* f3w  = (const float*)d_in[9];
  const float* f3b  = (const float*)d_in[10];
  const float* b1g = (const float*)d_in[11], *b1b = (const float*)d_in[12],
             * b1m = (const float*)d_in[13], *b1v = (const float*)d_in[14];
  const float* b2g = (const float*)d_in[15], *b2b = (const float*)d_in[16],
             * b2m = (const float*)d_in[17], *b2v = (const float*)d_in[18];
  const float* g1g = (const float*)d_in[19], *g1b = (const float*)d_in[20],
             * g1m = (const float*)d_in[21], *g1v = (const float*)d_in[22];
  const float* g2g = (const float*)d_in[23], *g2b = (const float*)d_in[24],
             * g2m = (const float*)d_in[25], *g2v = (const float*)d_in[26];
  const float* g3g = (const float*)d_in[27], *g3b = (const float*)d_in[28],
             * g3m = (const float*)d_in[29], *g3v = (const float*)d_in[30];

  char* ws = (char*)d_ws;
  unsigned char* act1 = (unsigned char*)(ws + ACT1_OFF);
  unsigned char* fc1i = (unsigned char*)(ws + FC1IN_OFF);
  float* cst          = (float*)(ws + CONST_OFF);
  unsigned int* bwp   = (unsigned int*)(ws + BPACK_OFF);

  prep_kernel<<<64, 256, 0, stream>>>(c1w, c1b, c2w, c2b, f1w, f1b, f2w, f2b, f3w, f3b,
                                      b1g, b1b, b1m, b1v, b2g, b2b, b2m, b2v,
                                      g1g, g1b, g1m, g1v, g2g, g2b, g2m, g2v,
                                      g3g, g3b, g3m, g3v, cst, bwp);
  conv1_kernel<<<9216, 128, 0, stream>>>(x, cst, bwp, act1);
  conv2_kernel<<<8192, 128, 0, stream>>>(act1, bwp, cst, fc1i);
  fc_kernel<<<128, 128, 0, stream>>>(fc1i, bwp, cst, (float*)d_out);
}
